// CWLSTM_27668179321363
// MI455X (gfx1250) — matrix operand order repaired and checked by reading, not compiled, not run on hardware
//
#include <hip/hip_runtime.h>

// ---------------------------------------------------------------------------
// Lattice-LSTM for MI455X (gfx1250, wave32).
// Phase A: TDM(tensor_load_to_lds) -> LDS double-buffer -> bf16 WMMA GEMMs
//          for the time-parallel projections px/pa/pw.
// Phase B: single persistent-WGP workgroup runs the 512-step lattice scan;
//          the per-step (M<=16)x768x768 skip-edge GEMM runs on WMMA as well.
// ---------------------------------------------------------------------------

typedef __bf16 bf16;
typedef __attribute__((ext_vector_type(16))) __bf16 v16bf;
typedef __attribute__((ext_vector_type(8)))  __bf16 v8bf;
typedef __attribute__((ext_vector_type(8)))  float  v8f;
typedef __attribute__((ext_vector_type(4)))  unsigned int v4u;
typedef __attribute__((ext_vector_type(8)))  int v8i_;
typedef __attribute__((ext_vector_type(4)))  int v4i_;

#define T_   512
#define K_   4
#define D_   768
#define H_   768
#define G_   2304      // 3*H
#define DW_  300
#define DWP_ 320       // DW padded to multiple of 32 for WMMA K
#define MMAX 16        // lattice fan-in bound: lens in [2,5] -> 4 srcs * K=4

#define BM 64          // GEMM macro-tile M
#define BN 128         // GEMM macro-tile N
#define BK 64          // GEMM K-step (elements)

#if defined(__has_builtin)
#if __has_builtin(__builtin_amdgcn_tensor_load_to_lds) && __has_builtin(__builtin_amdgcn_s_wait_tensorcnt)
#define HAVE_TDM 1
#endif
#endif

// ---------------- prep kernels ---------------------------------------------

__global__ void lat_cvt_bf16(bf16* __restrict__ dst, const float* __restrict__ src, int n) {
  int i = blockIdx.x * blockDim.x + threadIdx.x;
  if (i < n) dst[i] = (bf16)src[i];
}

// src[K][N] f32 (row-major) -> dst[N][Kp] bf16, zero-padded K..Kp
__global__ void lat_transpose_pad(bf16* __restrict__ dst, const float* __restrict__ src,
                                  int K, int N, int Kp) {
  int i = blockIdx.x * blockDim.x + threadIdx.x;
  if (i >= N * Kp) return;
  int n = i / Kp, k = i % Kp;
  dst[i] = (k < K) ? (bf16)src[(size_t)k * N + n] : (bf16)0.0f;
}

__global__ void lat_gather_emb(bf16* __restrict__ dst, const float* __restrict__ emb,
                               const int* __restrict__ word_ids) {
  int i = blockIdx.x * blockDim.x + threadIdx.x;
  if (i >= T_ * K_ * DWP_) return;
  int m = i / DWP_, k = i % DWP_;
  dst[i] = (k < DW_) ? (bf16)emb[(size_t)word_ids[m] * DW_ + k] : (bf16)0.0f;
}

// ---------------- Tensor Data Mover: 2-D tile -> LDS ------------------------
// Builds the D# per CDNA5 ISA and issues
// TENSOR_LOAD_TO_LDS. rows x BK bf16 tile, row stride = Kp elements.

#if HAVE_TDM
__device__ __forceinline__ void tdm_load_tile(unsigned lds_off, const bf16* gsrc,
                                              unsigned rows, unsigned kp,
                                              unsigned mrows) {
  unsigned long long ga = (unsigned long long)(const void*)gsrc;
  v4u g0;
  g0.x = 1u;                                  // count=1 (valid), user descriptor
  g0.y = lds_off;                             // lds_addr (byte offset)
  g0.z = (unsigned)(ga & 0xffffffffu);        // global_addr[31:0]
  g0.w = (unsigned)((ga >> 32) & 0x01ffffffu) | (2u << 30);  // addr[56:32] | type=2
  v8i_ g1;
  g1[0] = (int)(1u << 16);                    // wg_mask=0, data_size=1 (2 bytes)
  g1[1] = (int)((kp & 0xffffu) << 16);        // tensor_dim0[15:0] at bits 63:48
  g1[2] = (int)((kp >> 16) | ((mrows & 0xffffu) << 16));   // dim0 hi | dim1 lo
  g1[3] = (int)((mrows >> 16) | ((BK & 0xffffu) << 16));   // dim1 hi | tile_dim0
  g1[4] = (int)(rows & 0xffffu);              // tile_dim1 | tile_dim2=0
  g1[5] = (int)kp;                            // tensor_dim0_stride[31:0]
  g1[6] = 0;                                  // stride hi16 | dim1_stride lo16
  g1[7] = 0;
  v4i_ z = {0, 0, 0, 0};
#if defined(__clang_major__) && __clang_major__ >= 23
  v8i_ z8 = {0, 0, 0, 0, 0, 0, 0, 0};
  __builtin_amdgcn_tensor_load_to_lds(g0, g1, z, z, z8, 0);
#else
  __builtin_amdgcn_tensor_load_to_lds(g0, g1, z, z, 0);
#endif
}
#endif

// ---------------- WMMA GEMM: C[M][N] f32 = A[M][Kp] bf16 * Bt[N][Kp]^T ------
// Block = 256 threads (8 wave32). Macro-tile 64x128, each wave a 32x32
// sub-tile (2x2 fragments, 8 WMMAs per BK=64 step). Operands staged in LDS
// by TDM (double-buffered, wave0 issues, s_wait_tensorcnt gates).

__device__ __forceinline__ v8f wmma_bf16(v16bf a, v16bf b, v8f c) {
  return __builtin_amdgcn_wmma_f32_16x16x32_bf16(false, a, false, b, (short)0, c,
                                                 false, false);
}

__global__ void __launch_bounds__(256)
lat_wmma_gemm(const bf16* __restrict__ A, const bf16* __restrict__ Bt,
              float* __restrict__ C, int Mrows, int Ncols, int Kp) {
  __shared__ __attribute__((aligned(64))) bf16 sA[2][BM * BK];
  __shared__ __attribute__((aligned(64))) bf16 sB[2][BN * BK];
  const int tid  = threadIdx.x;
  const int lane = tid & 31, w = tid >> 5;
  const int wm = w >> 2, wn = w & 3;         // 2x4 wave grid
  const int h = lane >> 4, r = lane & 15;
  const int row0 = blockIdx.y * BM;
  const int col0 = blockIdx.x * BN;
  const int nK = Kp / BK;
  v8f acc[2][2] = {};

#if HAVE_TDM
  if (w == 0) {
    tdm_load_tile((unsigned)(unsigned long long)(void*)&sA[0][0],
                  A + (size_t)row0 * Kp, BM, (unsigned)Kp, (unsigned)Mrows);
    tdm_load_tile((unsigned)(unsigned long long)(void*)&sB[0][0],
                  Bt + (size_t)col0 * Kp, BN, (unsigned)Kp, (unsigned)Ncols);
  }
#endif
  for (int ks = 0; ks < nK; ++ks) {
    const int s = ks & 1;
#if HAVE_TDM
    __syncthreads();   // prior compute done -> buffer (ks+1)&1 reusable
    if (w == 0) {
      if (ks + 1 < nK) {
        tdm_load_tile((unsigned)(unsigned long long)(void*)&sA[(ks + 1) & 1][0],
                      A + (size_t)row0 * Kp + (ks + 1) * BK, BM, (unsigned)Kp,
                      (unsigned)Mrows);
        tdm_load_tile((unsigned)(unsigned long long)(void*)&sB[(ks + 1) & 1][0],
                      Bt + (size_t)col0 * Kp + (ks + 1) * BK, BN, (unsigned)Kp,
                      (unsigned)Ncols);
        __builtin_amdgcn_s_wait_tensorcnt(2);  // current buffer landed
      } else {
        __builtin_amdgcn_s_wait_tensorcnt(0);
      }
    }
    __syncthreads();   // current buffer visible to all waves
#else
    __syncthreads();
    for (int i2 = tid; i2 < BM * BK; i2 += 256)
      sA[s][i2] = A[(size_t)(row0 + i2 / BK) * Kp + ks * BK + (i2 % BK)];
    for (int i2 = tid; i2 < BN * BK; i2 += 256)
      sB[s][i2] = Bt[(size_t)(col0 + i2 / BK) * Kp + ks * BK + (i2 % BK)];
    __syncthreads();
#endif
    const bf16* aBase = &sA[s][(wm * 32 + r) * BK];
    const bf16* bBase = &sB[s][(wn * 32 + r) * BK];
#pragma unroll
    for (int ksub = 0; ksub < BK; ksub += 32) {
      // A 16x32 frag: contiguous K=16h+0..15 (matches B)
      union { v16bf v; v8bf p[2]; } a0, a1;
      a0.p[0] = *(const v8bf*)(aBase + ksub + 16 * h);
      a0.p[1] = *(const v8bf*)(aBase + ksub + 16 * h + 8);
      a1.p[0] = *(const v8bf*)(aBase + 16 * BK + ksub + 16 * h);
      a1.p[1] = *(const v8bf*)(aBase + 16 * BK + ksub + 16 * h + 8);
      // B 32x16 frag from N-major: contiguous K=16h+0..15
      v16bf b0 = *(const v16bf*)(bBase + ksub + 16 * h);
      v16bf b1 = *(const v16bf*)(bBase + 16 * BK + ksub + 16 * h);
      acc[0][0] = wmma_bf16(a0.v, b0, acc[0][0]);
      acc[0][1] = wmma_bf16(a0.v, b1, acc[0][1]);
      acc[1][0] = wmma_bf16(a1.v, b0, acc[1][0]);
      acc[1][1] = wmma_bf16(a1.v, b1, acc[1][1]);
    }
  }
#pragma unroll
  for (int mi = 0; mi < 2; ++mi)
#pragma unroll
    for (int ni = 0; ni < 2; ++ni) {
      float* crow = C + (size_t)(row0 + wm * 32 + mi * 16 + 8 * h) * Ncols +
                    col0 + wn * 32 + ni * 16 + r;
#pragma unroll
      for (int rr = 0; rr < 8; ++rr) crow[(size_t)rr * Ncols] = acc[mi][ni][rr];
    }
}

// ---------------- sequential lattice recurrence ----------------------------

__device__ __forceinline__ float sigf(float x) { return 1.0f / (1.0f + __expf(-x)); }

__global__ void __launch_bounds__(1024)
lat_recurrence(const float* __restrict__ px, const float* __restrict__ pa,
               const float* __restrict__ pw,
               const float* __restrict__ w_hh, const bf16* __restrict__ aw_hhT,
               const float* __restrict__ ww_hh,
               const float* __restrict__ b, const float* __restrict__ ab,
               const float* __restrict__ wb,
               const int* __restrict__ in_idx, const float* __restrict__ in_mask,
               const float* __restrict__ word_mask,
               float* __restrict__ c_store, float* __restrict__ alpha_g,
               float* __restrict__ out_h, float* __restrict__ out_c, int Mi) {
  extern __shared__ float smem[];
  float* h_lds = smem;                 // H
  float* c_lds = h_lds + H_;           // H
  float* hn    = c_lds + H_;           // H
  float* cn    = hn + H_;              // H
  float* gates = cn + H_;              // 3H
  float* iv    = gates + G_;           // H
  float* ov    = iv + H_;              // H
  float* gv    = ov + H_;              // H
  float* cin   = gv + H_;              // MMAX*H f32
  bf16*  cin_bf = (bf16*)(cin + MMAX * H_);  // MMAX*H bf16 (32B aligned)
  const int tid = threadIdx.x;
  if (Mi > MMAX) Mi = MMAX;

  for (int j = tid; j < H_; j += 1024) { h_lds[j] = 0.0f; c_lds[j] = 0.0f; }
  __syncthreads();

  for (int t = 0; t < T_; ++t) {
    // --- phase 1: gates = px[t] + b + h @ w_hh -----------------------------
    for (int j = tid; j < G_; j += 1024) {
      float acc = px[(size_t)t * G_ + j] + b[j];
      const float* wcol = w_hh + j;
      for (int d = 0; d < H_; ++d) acc = fmaf(h_lds[d], wcol[(size_t)d * G_], acc);
      gates[j] = acc;
    }
    __syncthreads();

    // --- phase 2: gate activations; phase 3: gather c_in (f32 + bf16) ------
    for (int j = tid; j < H_; j += 1024) {
      iv[j] = sigf(gates[j]);
      ov[j] = sigf(gates[H_ + j]);
      gv[j] = tanhf(gates[2 * H_ + j]);
    }
    for (int i = tid; i < MMAX * H_; i += 1024) {
      int m = i / H_, d = i - m * H_;
      float v = 0.0f;
      if (m < Mi) v = c_store[(size_t)in_idx[t * Mi + m] * H_ + d];
      cin[i] = v;
      cin_bf[i] = (bf16)v;
    }
    __syncthreads();

    // --- phase 4a: alpha_pre[16][H] = cin_bf @ aw_hh on WMMA ---------------
    {
      const int w = tid >> 5, lane = tid & 31;
      const int h = lane >> 4, r = lane & 15;
      const bf16* arow = cin_bf + r * H_;
      for (int tn = w; tn < H_ / 16; tn += 32) {
        v8f acc = {};
        const bf16* brow = aw_hhT + (size_t)(tn * 16 + r) * H_;
        for (int k0 = 0; k0 < H_; k0 += 32) {
          union { v16bf v; v8bf p[2]; } a;
          a.p[0] = *(const v8bf*)(arow + k0 + 16 * h);
          a.p[1] = *(const v8bf*)(arow + k0 + 16 * h + 8);
          v16bf bb = *(const v16bf*)(brow + k0 + 16 * h);
          acc = wmma_bf16(a.v, bb, acc);
        }
#pragma unroll
        for (int rr = 0; rr < 8; ++rr)
          alpha_g[(size_t)(8 * h + rr) * H_ + tn * 16 + r] = acc[rr];
      }
    }
    __syncthreads();

    // --- phase 4b: exp-softmax skip merge, c1/h1 ---------------------------
    if (tid < H_) {
      const int j = tid;
      const float pab = pa[(size_t)t * H_ + j] + ab[j];
      const float wi = __expf(iv[j]);
      float num = wi * gv[j];
      float den = wi;
      float msum = 0.0f;
      for (int m = 0; m < Mi; ++m) {
        const float msk = in_mask[t * Mi + m];
        msum += msk;
        const float al = sigf(alpha_g[(size_t)m * H_ + j] + pab);
        const float wa = __expf(al) * msk;
        den += wa;
        num = fmaf(wa, cin[m * H_ + j], num);
      }
      const float c1 = (msum > 0.0f) ? (num / den)
                                     : fmaf(iv[j], gv[j], (1.0f - iv[j]) * c_lds[j]);
      const float h1 = ov[j] * tanhf(c1);
      cn[j] = c1; hn[j] = h1;
      out_h[(size_t)t * H_ + j] = h1;
      out_c[(size_t)t * H_ + j] = c1;
    }
    __syncthreads();

    // --- phase 5: word gates pre-act = wb + h1 @ ww_hh ---------------------
    for (int j = tid; j < G_; j += 1024) {
      float acc = wb[j];
      const float* wcol = ww_hh + j;
      for (int d = 0; d < H_; ++d) acc = fmaf(hn[d], wcol[(size_t)d * G_], acc);
      gates[j] = acc;
    }
    __syncthreads();

    // --- phase 6: word-LSTM cells into c_store; roll h,c -------------------
    if (tid < H_) {
      const int j = tid;
      for (int k = 0; k < K_; ++k) {
        const size_t base = (size_t)(t * K_ + k) * G_;
        const float f2 = sigf(pw[base + j] + gates[j]);
        const float i2 = sigf(pw[base + H_ + j] + gates[H_ + j]);
        const float g2 = tanhf(pw[base + 2 * H_ + j] + gates[2 * H_ + j]);
        const float ct = fmaf(f2, cn[j], i2 * g2) * word_mask[t * K_ + k];
        c_store[(size_t)(t * K_ + k) * H_ + j] = ct;
      }
      h_lds[j] = hn[j]; c_lds[j] = cn[j];
    }
    __syncthreads();
  }
}

// ---------------- host launch ----------------------------------------------

extern "C" void kernel_launch(void* const* d_in, const int* in_sizes, int n_in,
                              void* d_out, int out_size, void* d_ws, size_t ws_size,
                              hipStream_t stream) {
  const float* x        = (const float*)d_in[0];
  const float* emb      = (const float*)d_in[1];
  const float* w_ih     = (const float*)d_in[2];
  const float* w_hh     = (const float*)d_in[3];
  const float* b        = (const float*)d_in[4];
  const float* aw_ih    = (const float*)d_in[5];
  const float* aw_hh    = (const float*)d_in[6];
  const float* ab       = (const float*)d_in[7];
  const float* ww_ih    = (const float*)d_in[8];
  const float* ww_hh    = (const float*)d_in[9];
  const float* wb       = (const float*)d_in[10];
  const int*   word_ids = (const int*)d_in[11];
  const float* word_msk = (const float*)d_in[12];
  const int*   in_idx   = (const int*)d_in[13];
  const float* in_mask  = (const float*)d_in[14];
  const int Mi = in_sizes[13] / T_;

  size_t off = 0;
  auto carve = [&](size_t bytes) -> void* {
    void* p = (char*)d_ws + off;
    off += (bytes + 255) & ~(size_t)255;
    return p;
  };
  bf16*  x_bf    = (bf16*)carve((size_t)T_ * D_ * 2);
  bf16*  w_ihT   = (bf16*)carve((size_t)G_ * D_ * 2);
  bf16*  aw_ihT  = (bf16*)carve((size_t)H_ * D_ * 2);
  bf16*  aw_hhT  = (bf16*)carve((size_t)H_ * H_ * 2);
  bf16*  ww_ihT  = (bf16*)carve((size_t)G_ * DWP_ * 2);
  bf16*  we_bf   = (bf16*)carve((size_t)T_ * K_ * DWP_ * 2);
  float* px      = (float*)carve((size_t)T_ * G_ * 4);
  float* pa      = (float*)carve((size_t)T_ * H_ * 4);
  float* pw      = (float*)carve((size_t)T_ * K_ * G_ * 4);
  float* c_store = (float*)carve((size_t)T_ * K_ * H_ * 4);
  float* alpha_g = (float*)carve((size_t)MMAX * H_ * 4);

  float* out_h = (float*)d_out;
  float* out_c = out_h + (size_t)T_ * H_;

  { int n = T_ * D_;
    lat_cvt_bf16<<<(n + 255) / 256, 256, 0, stream>>>(x_bf, x, n); }
  lat_transpose_pad<<<(G_ * D_ + 255) / 256, 256, 0, stream>>>(w_ihT,  w_ih,  D_,  G_, D_);
  lat_transpose_pad<<<(H_ * D_ + 255) / 256, 256, 0, stream>>>(aw_ihT, aw_ih, D_,  H_, D_);
  lat_transpose_pad<<<(H_ * H_ + 255) / 256, 256, 0, stream>>>(aw_hhT, aw_hh, H_,  H_, H_);
  lat_transpose_pad<<<(G_ * DWP_ + 255) / 256, 256, 0, stream>>>(ww_ihT, ww_ih, DW_, G_, DWP_);
  lat_gather_emb<<<(T_ * K_ * DWP_ + 255) / 256, 256, 0, stream>>>(we_bf, emb, word_ids);

  // --- TDM + WMMA projection GEMMs ---
  lat_wmma_gemm<<<dim3(G_ / BN, T_ / BM), 256, 0, stream>>>(x_bf, w_ihT, px, T_, G_, D_);
  lat_wmma_gemm<<<dim3(H_ / BN, T_ / BM), 256, 0, stream>>>(x_bf, aw_ihT, pa, T_, H_, D_);
  lat_wmma_gemm<<<dim3(G_ / BN, (T_ * K_) / BM), 256, 0, stream>>>(we_bf, ww_ihT, pw,
                                                                   T_ * K_, G_, DWP_);

  // --- sequential scan: one persistent workgroup ---
  const size_t smemBytes =
      (size_t)(4 * H_ + G_ + 3 * H_ + MMAX * H_) * sizeof(float) +
      (size_t)MMAX * H_ * sizeof(bf16);
  lat_recurrence<<<1, 1024, smemBytes, stream>>>(
      px, pa, pw, w_hh, aw_hhT, ww_hh, b, ab, wb,
      in_idx, in_mask, word_msk, c_store, alpha_g, out_h, out_c, Mi);
}
